// MultiQueryAttention_90331752169599
// MI455X (gfx1250) — matrix operand order repaired and checked by reading, not compiled, not run on hardware
//
#include <hip/hip_runtime.h>
#include <hip/hip_bf16.h>

// ---------------------------------------------------------------------------
// MQA forward for MI455X (gfx1250, wave32, WMMA 16x16x32 f16 -> f32 accum)
//   q = hs @ Wq[n] + bq[n]   (per head)          -> f16 [H][B][S][D]
//   k = hs @ Wk + bk                              -> f16 [B][S][D]
//   v = hs @ Wv + bv  (stored transposed)         -> f16 [B][D][S]
//   out = softmax(q k^T) v  (flash, streaming)    -> f32 [H][B][S][D]
// Flash iterates 64 keys/step (16 WMMAs) to amortize softmax VALU overhead.
// ---------------------------------------------------------------------------

#define HIDDEN 1024
#define NHEADS 16
#define HEADD  64
#define BATCH  4
#define SEQ    2048
#define NTOK   (BATCH * SEQ)   // 8192

typedef __attribute__((ext_vector_type(16))) _Float16 v16h;
typedef __attribute__((ext_vector_type(8)))  float    v8f;

union H16 { v16h v; _Float16 h[16]; uint4 q[2]; };

__device__ __forceinline__ v8f wmma_f32_f16(v16h a, v16h b, v8f c) {
  // D = A(16x32 f16) * B(32x16 f16) + C(16x16 f32)
  return __builtin_amdgcn_wmma_f32_16x16x32_f16(
      /*neg_a=*/false, a, /*neg_b=*/false, b,
      /*c_mod=*/(short)0, c, /*reuse_a=*/false, /*reuse_b=*/false);
}

// A fragment (16x32 f16). Lane L holds row m = L&15 (hi = L>>4):
//   elem i -> k = 16*hi + i (16 contiguous halves)
// rowp points at [row m, k 0] of the tile (row stride applied by caller).
__device__ __forceinline__ v16h frag_a(const _Float16* __restrict__ rowp, int hi) {
  H16 r;
  r.q[0] = *(const uint4*)(rowp + 16 * hi);
  r.q[1] = *(const uint4*)(rowp + 16 * hi + 8);
  return r.v;
}

// B fragment (32x16 f16) from K-major storage (column n contiguous in k).
// Lane L holds col n = L&15; elem i -> k = 16*hi + i (16 contiguous halves).
__device__ __forceinline__ v16h frag_b(const _Float16* __restrict__ colp, int hi) {
  H16 r;
  r.q[0] = *(const uint4*)(colp + 16 * hi);
  r.q[1] = *(const uint4*)(colp + 16 * hi + 8);
  return r.v;
}

// ------------------------- precision conversion ----------------------------

__global__ void cvt_f32_f16(const float* __restrict__ in,
                            _Float16* __restrict__ out, int n) {
  int i = blockIdx.x * blockDim.x + threadIdx.x;
  if (i < n) out[i] = (_Float16)in[i];
}

// in: [nmats][1024][64] f32 -> out: [nmats][64][1024] f16 (K-major for frag_b)
__global__ void cvt_w_transpose(const float* __restrict__ in,
                                _Float16* __restrict__ out, int total) {
  int e = blockIdx.x * blockDim.x + threadIdx.x;
  if (e >= total) return;
  int mat = e >> 16;        // 1024*64 elements per matrix
  int rem = e & 65535;
  int r = rem >> 6;         // hidden index (k)
  int c = rem & 63;         // head_dim index (n)
  out[(mat << 16) + (c << 10) + r] = (_Float16)in[e];
}

// ----------------------------- projections ---------------------------------
// Block = 128 threads = 4 waves; each wave computes a 16(M) x 64(N) strip.
// grid.x = NTOK/64 row tiles, grid.y = head index (1 for K/V).
__global__ void proj_gemm(const _Float16* __restrict__ Xh,   // [8192][1024]
                          const _Float16* __restrict__ Wt,   // [heads][64][1024]
                          const float*   __restrict__ bias,  // [heads][64] f32
                          _Float16*      __restrict__ out,
                          int transposed)                    // 0: [h][s][d], 1: [b][d][s]
{
  const int lane = threadIdx.x & 31;
  const int wave = threadIdx.x >> 5;
  const int hi = lane >> 4, m = lane & 15;
  const int head = blockIdx.y;
  const int row0 = (blockIdx.x * 4 + wave) * 16;

  const _Float16* wt   = Wt + (size_t)head * (HEADD * HIDDEN);
  const float*    bs   = bias + head * HEADD;
  const _Float16* arow = Xh + (size_t)(row0 + m) * HIDDEN;

  v8f acc[4] = {};
  for (int k = 0; k < HIDDEN; k += 32) {
    v16h a = frag_a(arow + k, hi);
#pragma unroll
    for (int nt = 0; nt < 4; ++nt) {
      v16h b = frag_b(wt + (size_t)(nt * 16 + m) * HIDDEN + k, hi);
      acc[nt] = wmma_f32_f16(a, b, acc[nt]);
    }
  }

#pragma unroll
  for (int nt = 0; nt < 4; ++nt) {
    const int col = nt * 16 + m;
    const float bval = bs[col];
#pragma unroll
    for (int r = 0; r < 8; ++r) {
      const int row = row0 + r + 8 * hi;         // global token index
      const float val = acc[nt][r] + bval;
      if (!transposed) {
        out[(size_t)head * NTOK * HEADD + (size_t)row * HEADD + col] = (_Float16)val;
      } else {
        const int bidx = row >> 11, s = row & (SEQ - 1);
        out[((size_t)bidx * HEADD + col) * SEQ + s] = (_Float16)val;
      }
    }
  }
}

// --------------------------- flash attention -------------------------------
// 256 threads = 8 waves per block; one wave per 16-query tile. All 8 waves of
// a block share the same (head, batch) K/V stream (consecutive query tiles),
// keeping the 512 KB per-batch K/V working set hot in WGP$/L2.
// Per iteration: 64 keys -> 8 score WMMAs, one fused online-softmax update,
// P staged through a private 2 KB LDS slice, 8 PV WMMAs.
__global__ void flash_attn(const _Float16* __restrict__ Qh,  // [16][4][2048][64]
                           const _Float16* __restrict__ Kh,  // [4][2048][64]
                           const _Float16* __restrict__ Vt,  // [4][64][2048]
                           float* __restrict__ out)          // [16][4][2048][64]
{
  __shared__ __align__(16) _Float16 ldsP[8][16 * 64];  // per-wave P staging

  const int lane = threadIdx.x & 31;
  const int wave = threadIdx.x >> 5;
  const int hi = lane >> 4, m = lane & 15;

  const int idx  = blockIdx.x * 8 + wave;   // 0..8191
  const int qt   = idx & 127;               // query tile within (head,b)
  const int nb   = idx >> 7;
  const int b    = nb & 3;
  const int head = nb >> 2;
  const int q0   = qt * 16;

  const _Float16* qbase = Qh + ((size_t)(head * BATCH + b) * SEQ + q0) * HEADD;
  const _Float16* kb    = Kh + (size_t)b * SEQ * HEADD;
  const _Float16* vb    = Vt + (size_t)b * HEADD * SEQ;

  // Q tile (16x64) as two A fragments, resident for the whole loop.
  v16h qa0 = frag_a(qbase + (size_t)m * HEADD, hi);
  v16h qa1 = frag_a(qbase + (size_t)m * HEADD + 32, hi);

  float mrow[8], lrow[8];
  v8f o[4] = {};
#pragma unroll
  for (int r = 0; r < 8; ++r) { mrow[r] = -3.0e38f; lrow[r] = 0.0f; }

  _Float16* pst = ldsP[wave];

  for (int t = 0; t < SEQ; t += 64) {
    if (t + 64 < SEQ) {   // hint next K/V tiles into cache (global_prefetch)
      __builtin_prefetch(kb + (size_t)(t + 64 + m) * HEADD, 0, 0);
      __builtin_prefetch(vb + (size_t)m * SEQ + t + 64, 0, 0);
    }

    // scores S = Q K^T for keys [t, t+64): four 16x16 tiles, K-dim = D = 64
    v8f s[4] = {};
#pragma unroll
    for (int kt = 0; kt < 4; ++kt) {
      const _Float16* kp = kb + (size_t)(t + kt * 16 + m) * HEADD;
      s[kt] = wmma_f32_f16(qa0, frag_b(kp,      hi), s[kt]);
      s[kt] = wmma_f32_f16(qa1, frag_b(kp + 32, hi), s[kt]);
    }

    // fused online softmax over all 64 keys. C-layout: lane holds row
    // (r + 8*hi), col (lane&15); xor masks 1/2/4/8 reduce across the 16
    // columns without crossing the wave32 half boundary.
    float scale[8];
#pragma unroll
    for (int r = 0; r < 8; ++r) {
      float rx = fmaxf(fmaxf(s[0][r], s[1][r]), fmaxf(s[2][r], s[3][r]));
      rx = fmaxf(rx, __shfl_xor(rx, 1, 32));
      rx = fmaxf(rx, __shfl_xor(rx, 2, 32));
      rx = fmaxf(rx, __shfl_xor(rx, 4, 32));
      rx = fmaxf(rx, __shfl_xor(rx, 8, 32));
      const float mn = fmaxf(mrow[r], rx);
      scale[r] = __expf(mrow[r] - mn);
      mrow[r]  = mn;
      float e0 = __expf(s[0][r] - mn);
      float e1 = __expf(s[1][r] - mn);
      float e2 = __expf(s[2][r] - mn);
      float e3 = __expf(s[3][r] - mn);
      float rs = (e0 + e1) + (e2 + e3);
      rs += __shfl_xor(rs, 1, 32);
      rs += __shfl_xor(rs, 2, 32);
      rs += __shfl_xor(rs, 4, 32);
      rs += __shfl_xor(rs, 8, 32);
      lrow[r] = lrow[r] * scale[r] + rs;
      // stage P (f16) row-major [16][64] for re-load in A-layout
      _Float16* pr = pst + (r + 8 * hi) * 64 + m;
      pr[0]  = (_Float16)e0;
      pr[16] = (_Float16)e1;
      pr[32] = (_Float16)e2;
      pr[48] = (_Float16)e3;
    }

#pragma unroll
    for (int nt = 0; nt < 4; ++nt)
#pragma unroll
      for (int r = 0; r < 8; ++r)
        o[nt][r] = o[nt][r] * scale[r];

    // O += P @ V ; P is two 16x32 A fragments, V tiles K-major in Vt[b][d][s]
#pragma unroll
    for (int half = 0; half < 2; ++half) {
      v16h pa = frag_a(pst + m * 64 + 32 * half, hi);
#pragma unroll
      for (int nt = 0; nt < 4; ++nt)
        o[nt] = wmma_f32_f16(
            pa, frag_b(vb + (size_t)(nt * 16 + m) * SEQ + t + 32 * half, hi),
            o[nt]);
    }
  }

  // epilogue: normalize by row sums, write f32 output [H][B][S][D]
  float* ob = out + ((size_t)(head * BATCH + b) * SEQ + q0) * HEADD;
#pragma unroll
  for (int r = 0; r < 8; ++r) {
    const float rinv = 1.0f / lrow[r];
#pragma unroll
    for (int nt = 0; nt < 4; ++nt)
      ob[(size_t)(r + 8 * hi) * HEADD + nt * 16 + m] = o[nt][r] * rinv;
  }
}

// ------------------------------ launcher -----------------------------------

extern "C" void kernel_launch(void* const* d_in, const int* in_sizes, int n_in,
                              void* d_out, int out_size, void* d_ws, size_t ws_size,
                              hipStream_t stream) {
  const float* hs = (const float*)d_in[0];  // [4][2048][1024]
  const float* Wq = (const float*)d_in[1];  // [16][1024][64]
  const float* bq = (const float*)d_in[2];  // [16][64]
  const float* Wk = (const float*)d_in[3];  // [1024][64]
  const float* bk = (const float*)d_in[4];  // [64]
  const float* Wv = (const float*)d_in[5];  // [1024][64]
  const float* bv = (const float*)d_in[6];  // [64]
  float* out = (float*)d_out;               // [16][4][2048][64]

  // workspace layout (~38 MB total)
  char* ws = (char*)d_ws;
  size_t off = 0;
  _Float16* hs_h = (_Float16*)(ws + off); off += (size_t)NTOK * HIDDEN * 2;           // 16.8 MB
  _Float16* wq_t = (_Float16*)(ws + off); off += (size_t)NHEADS * HEADD * HIDDEN * 2; // 2.1 MB
  _Float16* wk_t = (_Float16*)(ws + off); off += (size_t)HEADD * HIDDEN * 2;          // 128 KB
  _Float16* wv_t = (_Float16*)(ws + off); off += (size_t)HEADD * HIDDEN * 2;          // 128 KB
  _Float16* q_h  = (_Float16*)(ws + off); off += (size_t)NHEADS * NTOK * HEADD * 2;   // 16.8 MB
  _Float16* k_h  = (_Float16*)(ws + off); off += (size_t)NTOK * HEADD * 2;            // 1.0 MB
  _Float16* v_t  = (_Float16*)(ws + off); off += (size_t)NTOK * HEADD * 2;            // 1.0 MB

  { int n = NTOK * HIDDEN;
    cvt_f32_f16<<<(n + 255) / 256, 256, 0, stream>>>(hs, hs_h, n); }
  { int n = NHEADS * HIDDEN * HEADD;
    cvt_w_transpose<<<(n + 255) / 256, 256, 0, stream>>>(Wq, wq_t, n); }
  { int n = HIDDEN * HEADD;
    cvt_w_transpose<<<(n + 255) / 256, 256, 0, stream>>>(Wk, wk_t, n); }
  { int n = HIDDEN * HEADD;
    cvt_w_transpose<<<(n + 255) / 256, 256, 0, stream>>>(Wv, wv_t, n); }

  // projections: 128 threads (4 waves), each wave a 16x64 output strip
  proj_gemm<<<dim3(NTOK / 64, NHEADS), 128, 0, stream>>>(hs_h, wq_t, bq, q_h, 0);
  proj_gemm<<<dim3(NTOK / 64, 1),      128, 0, stream>>>(hs_h, wk_t, bk, k_h, 0);
  proj_gemm<<<dim3(NTOK / 64, 1),      128, 0, stream>>>(hs_h, wv_t, bv, v_t, 1);

  // flash attention: 8192 waves (one per 16-query tile), 8 waves per block
  flash_attn<<<NHEADS * BATCH * (SEQ / 16) / 8, 256, 0, stream>>>(q_h, k_h, v_t, out);

  (void)in_sizes; (void)n_in; (void)out_size; (void)ws_size;
}
